// VoxelTransformerLayer3D_84507776516326
// MI455X (gfx1250) — matrix operand order repaired and checked by reading, not compiled, not run on hardware
//
#include <hip/hip_runtime.h>
#include <cstdint>
#include <cstddef>

typedef __attribute__((ext_vector_type(16))) _Float16 v16h;
typedef __attribute__((ext_vector_type(8)))  _Float16 v8h;
typedef __attribute__((ext_vector_type(8)))  float    v8f;

#define LN_EPS 1e-5f

// -------------------- weight convert: W[Cin x Cout] f32 -> WT[Cout x Cin] f16 --------------------
__global__ void wt_transpose_kernel(const float* __restrict__ W, _Float16* __restrict__ WT,
                                    int Cin, int Cout) {
  int idx = blockIdx.x * blockDim.x + threadIdx.x;
  if (idx >= Cin * Cout) return;
  int k = idx % Cin;
  int n = idx / Cin;
  WT[idx] = (_Float16)W[(size_t)k * Cout + n];
}

// -------------------- layernorm: one block per row, blockDim.x == C (256) --------------------
__global__ void ln_kernel(const float* __restrict__ x, const float* __restrict__ g,
                          const float* __restrict__ b, _Float16* __restrict__ outh,
                          _Float16* __restrict__ rawh, int C) {
  int row = blockIdx.x;
  int t = threadIdx.x;
  const float* xr = x + (size_t)row * C;
  float v = xr[t];
  float s = v, s2 = v * v;
#pragma unroll
  for (int o = 16; o > 0; o >>= 1) {
    s  += __shfl_xor(s,  o, 32);
    s2 += __shfl_xor(s2, o, 32);
  }
  __shared__ float ss[8], ss2[8];
  int wid = t >> 5;
  if ((t & 31) == 0) { ss[wid] = s; ss2[wid] = s2; }
  __syncthreads();
  float tot = 0.f, tot2 = 0.f;
#pragma unroll
  for (int i = 0; i < 8; i++) { tot += ss[i]; tot2 += ss2[i]; }
  float mu  = tot / (float)C;
  float var = tot2 / (float)C - mu * mu;
  float inv = rsqrtf(var + LN_EPS);
  float y = (v - mu) * inv * g[t] + b[t];
  outh[(size_t)row * C + t] = (_Float16)y;
  if (rawh) rawh[(size_t)row * C + t] = (_Float16)v;
}

// -------------------- register-blocked WMMA GEMM --------------------
// Each wave computes a 64x32 output macro-tile: 4 M-tiles x 2 N-tiles, 8 f32 accumulators.
// Per K-step: 4 A-fragments + 2 B-fragments feed 8 independent v_wmma_f32_16x16x32_f16.
// Y = act(A[N x Cin] @ BT[Cout x Cin]^T + bias) (+ residual), f32 and/or f16 stores.
__global__ void gemm_wmma_kernel(const _Float16* __restrict__ A, const _Float16* __restrict__ BT,
                                 const float* __restrict__ bias, const float* __restrict__ residual,
                                 float* __restrict__ Yf, _Float16* __restrict__ Yh,
                                 int Nrows, int Cin, int Cout, int doGelu) {
  constexpr int MT = 4;   // 16-row tiles per wave
  constexpr int NTW = 2;  // 16-col tiles per wave
  int wavesPerBlock = blockDim.x >> 5;
  int wid = blockIdx.x * wavesPerBlock + ((int)threadIdx.x >> 5);
  int tilesN = Cout >> 5;                  // 32-col super-tiles
  int tilesTotal = (Nrows >> 6) * tilesN;  // 64-row super-tiles
  if (wid >= tilesTotal) return;  // whole-wave exit: EXEC stays all-ones for WMMA
  int tm = wid / tilesN;
  int tn = wid % tilesN;
  int lane = (int)threadIdx.x & 31;
  int hi   = lane >> 4;   // half-wave select
  int l15  = lane & 15;
  int row0 = tm << 6, col0 = tn << 5;

  // A fragment rows: lanes 0-15 cover K=[0..15], lanes 16-31 cover K=[16..31]
  const _Float16* arow[MT];
#pragma unroll
  for (int m = 0; m < MT; m++)
    arow[m] = A + (size_t)(row0 + m * 16 + l15) * Cin + (hi ? 16 : 0);
  // B fragment rows: lanes 0-15 cover K=[0..15], lanes 16-31 cover K=[16..31] (contiguous in WT)
  const _Float16* brow[NTW];
#pragma unroll
  for (int n = 0; n < NTW; n++)
    brow[n] = BT + (size_t)(col0 + n * 16 + l15) * Cin + (hi ? 16 : 0);

  v8f c[MT][NTW];
#pragma unroll
  for (int m = 0; m < MT; m++)
#pragma unroll
    for (int n = 0; n < NTW; n++) c[m][n] = (v8f){};

  for (int kc = 0; kc < Cin; kc += 32) {
    v16h a[MT];
#pragma unroll
    for (int m = 0; m < MT; m++) {
      a[m] = *(const v16h*)(arow[m] + kc);  // K = kbase + 0..15
    }
    v16h bf[NTW];
#pragma unroll
    for (int n = 0; n < NTW; n++) bf[n] = *(const v16h*)(brow[n] + kc);
#pragma unroll
    for (int m = 0; m < MT; m++)
#pragma unroll
      for (int n = 0; n < NTW; n++)
        c[m][n] = __builtin_amdgcn_wmma_f32_16x16x32_f16(false, a[m], false, bf[n],
                                                         (short)0, c[m][n], false, false);
  }

#pragma unroll
  for (int n = 0; n < NTW; n++) {
    int col = col0 + n * 16 + l15;
    float bn = bias ? bias[col] : 0.0f;
#pragma unroll
    for (int m = 0; m < MT; m++) {
#pragma unroll
      for (int r = 0; r < 8; r++) {
        int row = row0 + m * 16 + r + (hi << 3);  // lanes 16-31 hold rows M=8..15
        float v = c[m][n][r] + bn;
        if (doGelu) v = 0.5f * v * (1.0f + erff(v * 0.70710678118654752f));
        if (residual) v += residual[(size_t)row * Cout + col];
        if (Yf) Yf[(size_t)row * Cout + col] = v;
        if (Yh) Yh[(size_t)row * Cout + col] = (_Float16)v;
      }
    }
  }
}

// -------------------- windowed attention: one thread per (voxel, head) --------------------
__global__ void attn_kernel(const _Float16* __restrict__ Qh, const _Float16* __restrict__ Kh,
                            const _Float16* __restrict__ Vh, _Float16* __restrict__ aoh,
                            int total) {
  const int HD = 32;
  int gid = blockIdx.x * blockDim.x + threadIdx.x;
  if (gid >= total) return;
  int head = gid & 7;
  int vox  = gid >> 3;
  int w = vox & 15, h = (vox >> 4) & 15, d = (vox >> 8) & 15, b = vox >> 12;

  const _Float16* qp = Qh + (size_t)vox * 256 + head * HD;
  float q[HD];
#pragma unroll
  for (int e = 0; e < HD; e++) q[e] = (float)qp[e];

  float scores[27];
  float mx = -1e30f;
#pragma unroll
  for (int kk = 0; kk < 27; kk++) {
    int di = kk / 9 - 1, dj = (kk / 3) % 3 - 1, dk = kk % 3 - 1;
    int dd = d + di, hh = h + dj, ww = w + dk;
    bool valid = (kk != 13) && (unsigned)dd < 16u && (unsigned)hh < 16u && (unsigned)ww < 16u;
    float sc = -1e30f;
    if (valid) {
      int nb = ((b * 16 + dd) * 16 + hh) * 16 + ww;
      const _Float16* kp = Kh + (size_t)nb * 256 + head * HD;
      float acc = 0.f;
#pragma unroll
      for (int e = 0; e < HD; e++) acc += q[e] * (float)kp[e];
      sc = acc * 0.17677669529663687f;  // 1/sqrt(32)
    }
    scores[kk] = sc;
    mx = fmaxf(mx, sc);
  }

  float denom = 0.f;
#pragma unroll
  for (int kk = 0; kk < 27; kk++) {
    float e = (scores[kk] > -1e29f) ? __expf(scores[kk] - mx) : 0.0f;
    scores[kk] = e;
    denom += e;
  }
  float rden = (denom > 0.f) ? 1.0f / denom : 0.0f;

  float acc[HD];
#pragma unroll
  for (int e = 0; e < HD; e++) acc[e] = 0.f;
#pragma unroll
  for (int kk = 0; kk < 27; kk++) {
    float wgt = scores[kk] * rden;
    if (wgt != 0.0f) {
      int di = kk / 9 - 1, dj = (kk / 3) % 3 - 1, dk = kk % 3 - 1;
      int nb = ((b * 16 + (d + di)) * 16 + (h + dj)) * 16 + (w + dk);
      const _Float16* vp = Vh + (size_t)nb * 256 + head * HD;
#pragma unroll
      for (int e = 0; e < HD; e++) acc[e] += wgt * (float)vp[e];
    }
  }

  _Float16* op = aoh + (size_t)vox * 256 + head * HD;
#pragma unroll
  for (int e = 0; e < HD; e++) op[e] = (_Float16)acc[e];
}

// -------------------- host launch --------------------
extern "C" void kernel_launch(void* const* d_in, const int* in_sizes, int n_in,
                              void* d_out, int out_size, void* d_ws, size_t ws_size,
                              hipStream_t stream) {
  const float* x   = (const float*)d_in[0];
  const float* Wq  = (const float*)d_in[1];
  const float* bq  = (const float*)d_in[2];
  const float* Wk  = (const float*)d_in[3];
  const float* bk  = (const float*)d_in[4];
  const float* Wv  = (const float*)d_in[5];
  const float* bv  = (const float*)d_in[6];
  const float* Wo  = (const float*)d_in[7];
  const float* bo  = (const float*)d_in[8];
  const float* g1  = (const float*)d_in[9];
  const float* b1  = (const float*)d_in[10];
  const float* g2  = (const float*)d_in[11];
  const float* b2  = (const float*)d_in[12];
  const float* W1  = (const float*)d_in[13];
  const float* bf1 = (const float*)d_in[14];
  const float* W2  = (const float*)d_in[15];
  const float* bf2 = (const float*)d_in[16];

  const int C = 256, Ff = 1024;
  const int NT = in_sizes[0] / C;   // 8192 tokens

  char* ws = (char*)d_ws;
  size_t off = 0;
  auto walloc = [&](size_t bytes) -> void* {
    off = (off + 255) & ~(size_t)255;
    void* p = ws + off;
    off += bytes;
    return p;
  };

  _Float16* xh   = (_Float16*)walloc((size_t)NT * C * 2);
  _Float16* tnh  = (_Float16*)walloc((size_t)NT * C * 2);
  _Float16* Qh   = (_Float16*)walloc((size_t)NT * C * 2);
  _Float16* Kh   = (_Float16*)walloc((size_t)NT * C * 2);
  _Float16* Vh   = (_Float16*)walloc((size_t)NT * C * 2);
  _Float16* aoh  = (_Float16*)walloc((size_t)NT * C * 2);
  float*    tbuf = (float*)   walloc((size_t)NT * C * 4);
  _Float16* uh   = (_Float16*)walloc((size_t)NT * C * 2);
  _Float16* h1h  = (_Float16*)walloc((size_t)NT * Ff * 2);
  _Float16* WqT  = (_Float16*)walloc((size_t)C * C * 2);
  _Float16* WkT  = (_Float16*)walloc((size_t)C * C * 2);
  _Float16* WvT  = (_Float16*)walloc((size_t)C * C * 2);
  _Float16* WoT  = (_Float16*)walloc((size_t)C * C * 2);
  _Float16* W1T  = (_Float16*)walloc((size_t)C * Ff * 2);
  _Float16* W2T  = (_Float16*)walloc((size_t)Ff * C * 2);
  (void)ws_size;

  // 1) weight conversion / transpose
  {
    int n1 = C * C;
    wt_transpose_kernel<<<(n1 + 255) / 256, 256, 0, stream>>>(Wq, WqT, C, C);
    wt_transpose_kernel<<<(n1 + 255) / 256, 256, 0, stream>>>(Wk, WkT, C, C);
    wt_transpose_kernel<<<(n1 + 255) / 256, 256, 0, stream>>>(Wv, WvT, C, C);
    wt_transpose_kernel<<<(n1 + 255) / 256, 256, 0, stream>>>(Wo, WoT, C, C);
    int n2 = C * Ff;
    wt_transpose_kernel<<<(n2 + 255) / 256, 256, 0, stream>>>(W1, W1T, C, Ff);
    wt_transpose_kernel<<<(n2 + 255) / 256, 256, 0, stream>>>(W2, W2T, Ff, C);
  }

  // 2) LN1 (tn) + f16 copy of x
  ln_kernel<<<NT, C, 0, stream>>>(x, g1, b1, tnh, xh, C);

  // waves = (rows/64) * (cols/32); 8 waves per 256-thread block
  auto gemm_blocks = [](int rows, int cols) {
    int waves = (rows >> 6) * (cols >> 5);
    return (waves + 7) / 8;
  };

  // 3-5) Q = tn@Wq+bq ; K = x@Wk+bk ; V = x@Wv+bv   (all f16 out)
  {
    int blocks = gemm_blocks(NT, C);
    gemm_wmma_kernel<<<blocks, 256, 0, stream>>>(tnh, WqT, bq, nullptr, nullptr, Qh, NT, C, C, 0);
    gemm_wmma_kernel<<<blocks, 256, 0, stream>>>(xh,  WkT, bk, nullptr, nullptr, Kh, NT, C, C, 0);
    gemm_wmma_kernel<<<blocks, 256, 0, stream>>>(xh,  WvT, bv, nullptr, nullptr, Vh, NT, C, C, 0);
  }

  // 6) windowed attention -> aoh
  {
    int total = NT * 8;
    attn_kernel<<<(total + 255) / 256, 256, 0, stream>>>(Qh, Kh, Vh, aoh, total);
  }

  // 7) t = x + ao@Wo + bo   (f32)
  gemm_wmma_kernel<<<gemm_blocks(NT, C), 256, 0, stream>>>(aoh, WoT, bo, x, tbuf, nullptr,
                                                           NT, C, C, 0);

  // 8) LN2(t) -> uh
  ln_kernel<<<NT, C, 0, stream>>>(tbuf, g2, b2, uh, nullptr, C);

  // 9) h1 = gelu(u@W1 + bf1)  (f16, N x 1024)
  gemm_wmma_kernel<<<gemm_blocks(NT, Ff), 256, 0, stream>>>(uh, W1T, bf1, nullptr, nullptr, h1h,
                                                            NT, C, Ff, 1);

  // 10) out = t + h1@W2 + bf2  (f32 -> d_out)
  gemm_wmma_kernel<<<gemm_blocks(NT, C), 256, 0, stream>>>(h1h, W2T, bf2, tbuf, (float*)d_out,
                                                           nullptr, NT, Ff, C, 0);
}
